// UpdateBlock_901943132402
// MI455X (gfx1250) — matrix operand order repaired and checked by reading, not compiled, not run on hardware
//
#include <hip/hip_runtime.h>

// ---------------------------------------------------------------------------
// EGNN forward for MI455X (gfx1250): bf16 WMMA (16x16x32) for all GEMMs,
// fully fused per-edge MLP with atomic scatter aggregation.
// Round 4: software-pipelined K-loop + sched_barrier(0) between the prefetch
// loads and the WMMA chain so the scheduler cannot re-serialize them.
// ---------------------------------------------------------------------------

typedef __attribute__((ext_vector_type(16))) __bf16       v16bf;
typedef __attribute__((ext_vector_type(8)))  float        v8f;
typedef __attribute__((ext_vector_type(4)))  unsigned int u32x4;

union ABfrag { v16bf v; u32x4 q[2]; };

__device__ __forceinline__ void sched_fence() {
#if __has_builtin(__builtin_amdgcn_sched_barrier)
  __builtin_amdgcn_sched_barrier(0);
#endif
}

__device__ __forceinline__ unsigned short f32_to_bf16(float f) {
  unsigned int u = __float_as_uint(f);
  u += 0x7FFFu + ((u >> 16) & 1u);        // round-to-nearest-even
  return (unsigned short)(u >> 16);
}
__device__ __forceinline__ float sigmoidf_(float x) { return 1.0f / (1.0f + __expf(-x)); }
__device__ __forceinline__ float siluf_(float x)    { return x / (1.0f + __expf(-x)); }

__device__ __forceinline__ v8f wmma_bf16(v16bf a, v16bf b, v8f c) {
  return __builtin_amdgcn_wmma_f32_16x16x32_bf16(false, a, false, b, (short)0, c, false, false);
}

// Load the 4 B fragments (one per N-tile) for one 32-wide K-step.
__device__ __forceinline__ void load_b4(const unsigned short* wp, size_t stride,
                                        ABfrag* b) {
  b[0].q[0] = *(const u32x4*)(wp);
  b[0].q[1] = *(const u32x4*)(wp + 8);
  b[1].q[0] = *(const u32x4*)(wp + 16 * stride);
  b[1].q[1] = *(const u32x4*)(wp + 16 * stride + 8);
  b[2].q[0] = *(const u32x4*)(wp + 32 * stride);
  b[2].q[1] = *(const u32x4*)(wp + 32 * stride + 8);
  b[3].q[0] = *(const u32x4*)(wp + 48 * stride);
  b[3].q[1] = *(const u32x4*)(wp + 48 * stride + 8);
}

// A-fragment K-chunk (8 contiguous bf16 = 16B) for the edge MLP input
// inp = [h[row](0..127) | h[col](128..255) | radial,ea,0..(256..287)]
__device__ __forceinline__ u32x4 edge_chunk(const unsigned short* hr,
                                            const unsigned short* hc,
                                            unsigned int tail0, int c) {
  if (c < 16) return *(const u32x4*)(hr + c * 8);
  if (c < 32) return *(const u32x4*)(hc + (c - 16) * 8);
  if (c == 32) { u32x4 r = {tail0, 0u, 0u, 0u}; return r; }
  u32x4 z = {0u, 0u, 0u, 0u}; return z;
}

__device__ __forceinline__ void load_a_edge(ABfrag& a, const unsigned short* hr,
                                            const unsigned short* hc,
                                            unsigned int tail0, int ks, int hi) {
  a.q[0] = edge_chunk(hr, hc, tail0, ks * 4 + 2 * hi);
  a.q[1] = edge_chunk(hr, hc, tail0, ks * 4 + 2 * hi + 1);
}

// A-fragment K-chunk for node update input: [h_bf16(0..127) | agg_f32(128..255)]
__device__ __forceinline__ u32x4 node_chunk(const unsigned short* hr,
                                            const float* ar, int c) {
  if (c < 16) return *(const u32x4*)(hr + c * 8);
  const float* p = ar + (c - 16) * 8;
  unsigned int w0 = (unsigned)f32_to_bf16(p[0]) | ((unsigned)f32_to_bf16(p[1]) << 16);
  unsigned int w1 = (unsigned)f32_to_bf16(p[2]) | ((unsigned)f32_to_bf16(p[3]) << 16);
  unsigned int w2 = (unsigned)f32_to_bf16(p[4]) | ((unsigned)f32_to_bf16(p[5]) << 16);
  unsigned int w3 = (unsigned)f32_to_bf16(p[6]) | ((unsigned)f32_to_bf16(p[7]) << 16);
  u32x4 r = {w0, w1, w2, w3}; return r;
}

__device__ __forceinline__ void load_a_node(ABfrag& a, const unsigned short* hr,
                                            const float* ar, int ks, int hi) {
  a.q[0] = node_chunk(hr, ar, ks * 4 + 2 * hi);
  a.q[1] = node_chunk(hr, ar, ks * 4 + 2 * hi + 1);
}

__device__ __forceinline__ void load_a_lds(ABfrag& a, const unsigned short* tile,
                                           int ml, int ks, int hi) {
  a.q[0] = *(const u32x4*)&tile[ml * 136 + ks * 32 + hi * 16];
  a.q[1] = *(const u32x4*)&tile[ml * 136 + ks * 32 + hi * 16 + 8];
}

// ---------------------------------------------------------------------------
// Small prep kernels
// ---------------------------------------------------------------------------
__global__ void egnn_init_h(const float* __restrict__ h, float* __restrict__ hf,
                            unsigned short* __restrict__ hb, int n) {
  int i = blockIdx.x * blockDim.x + threadIdx.x;
  if (i >= n) return;
  float v = h[i];
  hf[i] = v;
  hb[i] = f32_to_bf16(v);
}

__global__ void egnn_geom(const float* __restrict__ x, const int* __restrict__ rows,
                          const int* __restrict__ cols, float* __restrict__ radial,
                          float* __restrict__ cdiff, int E) {
  int e = blockIdx.x * blockDim.x + threadIdx.x;
  if (e >= E) return;
  int r = rows[e], c = cols[e];
  float dx = x[r * 3 + 0] - x[c * 3 + 0];
  float dy = x[r * 3 + 1] - x[c * 3 + 1];
  float dz = x[r * 3 + 2] - x[c * 3 + 2];
  float rad = dx * dx + dy * dy + dz * dz;
  radial[e] = rad;
  float inv = 1.0f / (sqrtf(rad + 1e-8f) + 1.0f);   // NORM_CONSTANT = 1
  cdiff[e * 3 + 0] = dx * inv;
  cdiff[e * 3 + 1] = dy * inv;
  cdiff[e * 3 + 2] = dz * inv;
}

// W [K x 128] f32  ->  Wt [128 x Kp] bf16 (transposed, K zero-padded to Kp)
__global__ void egnn_conv_w(const float* __restrict__ W, unsigned short* __restrict__ Wt,
                            int K, int Kp) {
  int i = blockIdx.x * blockDim.x + threadIdx.x;
  if (i >= 128 * Kp) return;
  int n = i / Kp, k = i % Kp;
  Wt[i] = (k < K) ? f32_to_bf16(W[(size_t)k * 128 + n]) : (unsigned short)0;
}

__global__ void egnn_copy_f32(const float* __restrict__ src, float* __restrict__ dst, int n) {
  int i = blockIdx.x * blockDim.x + threadIdx.x;
  if (i < n) dst[i] = src[i];
}

__global__ void egnn_x_final(const float* __restrict__ x, const float* __restrict__ xagg,
                             float* __restrict__ out, int n) {
  int i = blockIdx.x * blockDim.x + threadIdx.x;
  if (i < n) out[i] = x[i] + xagg[i];
}

// ---------------------------------------------------------------------------
// Fused edge MLP (feature layers): m = silu(silu(inp@W1+b1)@W2+b2);
// att = sigmoid(m@aW+ab); scatter agg[row] += m*att/100
// ---------------------------------------------------------------------------
__global__ __launch_bounds__(256) void egnn_edge_feat(
    const unsigned short* __restrict__ hb,
    const int* __restrict__ rows, const int* __restrict__ cols,
    const float* __restrict__ radial, const float* __restrict__ eattr,
    const unsigned short* __restrict__ W1t, const float* __restrict__ b1,
    const unsigned short* __restrict__ W2t, const float* __restrict__ b2,
    const float* __restrict__ aW, const float* __restrict__ abv,
    float* __restrict__ agg, int E) {
  __shared__ unsigned short m1s[64 * 136];   // padded stride: bank-conflict free
  __shared__ float attb[64];
  __shared__ int rows_s[64];

  const int tid = threadIdx.x;
  const long eb = (long)blockIdx.x * 64;
  if (tid < 64) {
    long e = eb + tid; if (e >= E) e = E - 1;
    rows_s[tid] = rows[e];
    attb[tid] = abv[0];
  }
  __syncthreads();

  const int lane = tid & 31, wave = tid >> 5;
  const int mt = wave & 3, nh = wave >> 2;
  const int hi = lane >> 4, lm = lane & 15;
  const int nbase = nh * 64;

  long eA = eb + mt * 16 + lm; if (eA >= E) eA = E - 1;
  const unsigned short* hr = hb + (size_t)rows[eA] * 128;
  const unsigned short* hc = hb + (size_t)cols[eA] * 128;
  const unsigned int tail0 =
      (unsigned)f32_to_bf16(radial[eA]) | ((unsigned)f32_to_bf16(eattr[eA]) << 16);

  const unsigned short* w1base = W1t + (size_t)(nbase + lm) * 288 + hi * 16;
  const unsigned short* w2base = W2t + (size_t)(nbase + lm) * 128 + hi * 16;

  // -------- GEMM1: [16 x 288] @ [288 x 128] (this wave: 16x64 slice) --------
  v8f acc[4];
  #pragma unroll
  for (int nt = 0; nt < 4; ++nt) {
    float bv = b1[nbase + nt * 16 + lm];
    v8f t = {bv, bv, bv, bv, bv, bv, bv, bv};
    acc[nt] = t;
  }
  {
    ABfrag a[2], bf[2][4];
    load_a_edge(a[0], hr, hc, tail0, 0, hi);
    load_b4(w1base, 288, bf[0]);
    #pragma unroll
    for (int ks = 0; ks < 9; ++ks) {
      const int cur = ks & 1, nxt = cur ^ 1;
      if (ks + 1 < 9) {
        load_a_edge(a[nxt], hr, hc, tail0, ks + 1, hi);
        load_b4(w1base + (ks + 1) * 32, 288, bf[nxt]);
      }
      sched_fence();   // keep next-step loads above this step's WMMAs
      acc[0] = wmma_bf16(a[cur].v, bf[cur][0].v, acc[0]);
      acc[1] = wmma_bf16(a[cur].v, bf[cur][1].v, acc[1]);
      acc[2] = wmma_bf16(a[cur].v, bf[cur][2].v, acc[2]);
      acc[3] = wmma_bf16(a[cur].v, bf[cur][3].v, acc[3]);
      sched_fence();
    }
  }
  // silu -> LDS (bf16) for GEMM2 A-fragments
  #pragma unroll
  for (int nt = 0; nt < 4; ++nt)
    #pragma unroll
    for (int v = 0; v < 8; ++v)
      m1s[(mt * 16 + hi * 8 + v) * 136 + nbase + nt * 16 + lm] =
          f32_to_bf16(siluf_(acc[nt][v]));
  __syncthreads();

  // -------- GEMM2: [16 x 128] @ [128 x 128] --------
  v8f acc2[4];
  #pragma unroll
  for (int nt = 0; nt < 4; ++nt) {
    float bv = b2[nbase + nt * 16 + lm];
    v8f t = {bv, bv, bv, bv, bv, bv, bv, bv};
    acc2[nt] = t;
  }
  const int ml = mt * 16 + lm;
  {
    ABfrag a[2], bf[2][4];
    load_a_lds(a[0], m1s, ml, 0, hi);
    load_b4(w2base, 128, bf[0]);
    #pragma unroll
    for (int ks = 0; ks < 4; ++ks) {
      const int cur = ks & 1, nxt = cur ^ 1;
      if (ks + 1 < 4) {
        load_a_lds(a[nxt], m1s, ml, ks + 1, hi);
        load_b4(w2base + (ks + 1) * 32, 128, bf[nxt]);
      }
      sched_fence();
      acc2[0] = wmma_bf16(a[cur].v, bf[cur][0].v, acc2[0]);
      acc2[1] = wmma_bf16(a[cur].v, bf[cur][1].v, acc2[1]);
      acc2[2] = wmma_bf16(a[cur].v, bf[cur][2].v, acc2[2]);
      acc2[3] = wmma_bf16(a[cur].v, bf[cur][3].v, acc2[3]);
      sched_fence();
    }
  }
  float m2v[4][8];
  #pragma unroll
  for (int nt = 0; nt < 4; ++nt)
    #pragma unroll
    for (int v = 0; v < 8; ++v) m2v[nt][v] = siluf_(acc2[nt][v]);

  // -------- attention: att[m] = sigmoid(sum_n m2[m][n]*aW[n] + ab) --------
  float aWl[4];
  #pragma unroll
  for (int nt = 0; nt < 4; ++nt) aWl[nt] = aW[nbase + nt * 16 + lm];
  #pragma unroll
  for (int v = 0; v < 8; ++v) {
    float p = 0.f;
    #pragma unroll
    for (int nt = 0; nt < 4; ++nt) p += m2v[nt][v] * aWl[nt];
    #pragma unroll
    for (int off = 1; off < 16; off <<= 1) p += __shfl_xor(p, off);
    if (lm == 0) atomicAdd(&attb[mt * 16 + hi * 8 + v], p);
  }
  __syncthreads();

  // -------- scatter: agg[row] += m2 * att / 100 --------
  #pragma unroll
  for (int v = 0; v < 8; ++v) {
    long e = eb + mt * 16 + hi * 8 + v;
    if (e < E) {
      float sc = sigmoidf_(attb[mt * 16 + hi * 8 + v]) * 0.01f;
      float* ap = agg + (size_t)rows_s[mt * 16 + hi * 8 + v] * 128 + nbase + lm;
      #pragma unroll
      for (int nt = 0; nt < 4; ++nt) atomicAdd(ap + nt * 16, m2v[nt][v] * sc);
    }
  }
}

// ---------------------------------------------------------------------------
// Fused coord head: s = silu(silu(inp@cW1+cb1)@cW2+cb2)@cW3;
// xagg[row] += coord_diff * tanh(s) * 15 / 100
// ---------------------------------------------------------------------------
__global__ __launch_bounds__(256) void egnn_edge_coord(
    const unsigned short* __restrict__ hb,
    const int* __restrict__ rows, const int* __restrict__ cols,
    const float* __restrict__ radial, const float* __restrict__ eattr,
    const unsigned short* __restrict__ W1t, const float* __restrict__ b1,
    const unsigned short* __restrict__ W2t, const float* __restrict__ b2,
    const float* __restrict__ cW3, const float* __restrict__ cdiff,
    float* __restrict__ xagg, int E) {
  __shared__ unsigned short m1s[64 * 136];
  __shared__ float sb[64];
  __shared__ int rows_s[64];

  const int tid = threadIdx.x;
  const long eb = (long)blockIdx.x * 64;
  if (tid < 64) {
    long e = eb + tid; if (e >= E) e = E - 1;
    rows_s[tid] = rows[e];
    sb[tid] = 0.f;
  }
  __syncthreads();

  const int lane = tid & 31, wave = tid >> 5;
  const int mt = wave & 3, nh = wave >> 2;
  const int hi = lane >> 4, lm = lane & 15;
  const int nbase = nh * 64;

  long eA = eb + mt * 16 + lm; if (eA >= E) eA = E - 1;
  const unsigned short* hr = hb + (size_t)rows[eA] * 128;
  const unsigned short* hc = hb + (size_t)cols[eA] * 128;
  const unsigned int tail0 =
      (unsigned)f32_to_bf16(radial[eA]) | ((unsigned)f32_to_bf16(eattr[eA]) << 16);

  const unsigned short* w1base = W1t + (size_t)(nbase + lm) * 288 + hi * 16;
  const unsigned short* w2base = W2t + (size_t)(nbase + lm) * 128 + hi * 16;

  v8f acc[4];
  #pragma unroll
  for (int nt = 0; nt < 4; ++nt) {
    float bv = b1[nbase + nt * 16 + lm];
    v8f t = {bv, bv, bv, bv, bv, bv, bv, bv};
    acc[nt] = t;
  }
  {
    ABfrag a[2], bf[2][4];
    load_a_edge(a[0], hr, hc, tail0, 0, hi);
    load_b4(w1base, 288, bf[0]);
    #pragma unroll
    for (int ks = 0; ks < 9; ++ks) {
      const int cur = ks & 1, nxt = cur ^ 1;
      if (ks + 1 < 9) {
        load_a_edge(a[nxt], hr, hc, tail0, ks + 1, hi);
        load_b4(w1base + (ks + 1) * 32, 288, bf[nxt]);
      }
      sched_fence();
      acc[0] = wmma_bf16(a[cur].v, bf[cur][0].v, acc[0]);
      acc[1] = wmma_bf16(a[cur].v, bf[cur][1].v, acc[1]);
      acc[2] = wmma_bf16(a[cur].v, bf[cur][2].v, acc[2]);
      acc[3] = wmma_bf16(a[cur].v, bf[cur][3].v, acc[3]);
      sched_fence();
    }
  }
  #pragma unroll
  for (int nt = 0; nt < 4; ++nt)
    #pragma unroll
    for (int v = 0; v < 8; ++v)
      m1s[(mt * 16 + hi * 8 + v) * 136 + nbase + nt * 16 + lm] =
          f32_to_bf16(siluf_(acc[nt][v]));
  __syncthreads();

  v8f acc2[4];
  #pragma unroll
  for (int nt = 0; nt < 4; ++nt) {
    float bv = b2[nbase + nt * 16 + lm];
    v8f t = {bv, bv, bv, bv, bv, bv, bv, bv};
    acc2[nt] = t;
  }
  const int ml = mt * 16 + lm;
  {
    ABfrag a[2], bf[2][4];
    load_a_lds(a[0], m1s, ml, 0, hi);
    load_b4(w2base, 128, bf[0]);
    #pragma unroll
    for (int ks = 0; ks < 4; ++ks) {
      const int cur = ks & 1, nxt = cur ^ 1;
      if (ks + 1 < 4) {
        load_a_lds(a[nxt], m1s, ml, ks + 1, hi);
        load_b4(w2base + (ks + 1) * 32, 128, bf[nxt]);
      }
      sched_fence();
      acc2[0] = wmma_bf16(a[cur].v, bf[cur][0].v, acc2[0]);
      acc2[1] = wmma_bf16(a[cur].v, bf[cur][1].v, acc2[1]);
      acc2[2] = wmma_bf16(a[cur].v, bf[cur][2].v, acc2[2]);
      acc2[3] = wmma_bf16(a[cur].v, bf[cur][3].v, acc2[3]);
      sched_fence();
    }
  }

  // s[m] = sum_n silu(m2[m][n]) * cW3[n]
  float wl[4];
  #pragma unroll
  for (int nt = 0; nt < 4; ++nt) wl[nt] = cW3[nbase + nt * 16 + lm];
  #pragma unroll
  for (int v = 0; v < 8; ++v) {
    float p = 0.f;
    #pragma unroll
    for (int nt = 0; nt < 4; ++nt) p += siluf_(acc2[nt][v]) * wl[nt];
    #pragma unroll
    for (int off = 1; off < 16; off <<= 1) p += __shfl_xor(p, off);
    if (lm == 0) atomicAdd(&sb[mt * 16 + hi * 8 + v], p);
  }
  __syncthreads();

  if (tid < 64) {
    long e = eb + tid;
    if (e < E) {
      float f = tanhf(sb[tid]) * 15.0f * 0.01f;   // COORDS_RANGE / NORM_FACTOR
      int r = rows_s[tid];
      const float* cd = cdiff + (size_t)e * 3;
      atomicAdd(&xagg[r * 3 + 0], cd[0] * f);
      atomicAdd(&xagg[r * 3 + 1], cd[1] * f);
      atomicAdd(&xagg[r * 3 + 2], cd[2] * f);
    }
  }
}

// ---------------------------------------------------------------------------
// Node update: h += silu([h,agg]@uW1+ub1)@uW2+ub2  (in-place f32 + bf16 mirror)
// ---------------------------------------------------------------------------
__global__ __launch_bounds__(256) void egnn_node(
    const float* __restrict__ agg,
    const unsigned short* __restrict__ U1t, const float* __restrict__ ub1,
    const unsigned short* __restrict__ U2t, const float* __restrict__ ub2,
    float* hf, unsigned short* hbv, int N) {
  __shared__ unsigned short us[64 * 136];

  const int tid = threadIdx.x;
  const long nb0 = (long)blockIdx.x * 64;
  const int lane = tid & 31, wave = tid >> 5;
  const int mt = wave & 3, nh = wave >> 2;
  const int hi = lane >> 4, lm = lane & 15;
  const int nbase = nh * 64;

  long nd = nb0 + mt * 16 + lm; if (nd >= N) nd = N - 1;
  const unsigned short* hr = hbv + (size_t)nd * 128;
  const float* ar = agg + (size_t)nd * 128;

  const unsigned short* u1base = U1t + (size_t)(nbase + lm) * 256 + hi * 16;
  const unsigned short* u2base = U2t + (size_t)(nbase + lm) * 128 + hi * 16;

  // GEMM1: [16 x 256] @ [256 x 128]
  v8f acc[4];
  #pragma unroll
  for (int nt = 0; nt < 4; ++nt) {
    float bv = ub1[nbase + nt * 16 + lm];
    v8f t = {bv, bv, bv, bv, bv, bv, bv, bv};
    acc[nt] = t;
  }
  {
    ABfrag a[2], bf[2][4];
    load_a_node(a[0], hr, ar, 0, hi);
    load_b4(u1base, 256, bf[0]);
    #pragma unroll
    for (int ks = 0; ks < 8; ++ks) {
      const int cur = ks & 1, nxt = cur ^ 1;
      if (ks + 1 < 8) {
        load_a_node(a[nxt], hr, ar, ks + 1, hi);
        load_b4(u1base + (ks + 1) * 32, 256, bf[nxt]);
      }
      sched_fence();
      acc[0] = wmma_bf16(a[cur].v, bf[cur][0].v, acc[0]);
      acc[1] = wmma_bf16(a[cur].v, bf[cur][1].v, acc[1]);
      acc[2] = wmma_bf16(a[cur].v, bf[cur][2].v, acc[2]);
      acc[3] = wmma_bf16(a[cur].v, bf[cur][3].v, acc[3]);
      sched_fence();
    }
  }
  #pragma unroll
  for (int nt = 0; nt < 4; ++nt)
    #pragma unroll
    for (int v = 0; v < 8; ++v)
      us[(mt * 16 + hi * 8 + v) * 136 + nbase + nt * 16 + lm] =
          f32_to_bf16(siluf_(acc[nt][v]));
  __syncthreads();

  // GEMM2: [16 x 128] @ [128 x 128]
  v8f acc2[4];
  #pragma unroll
  for (int nt = 0; nt < 4; ++nt) {
    float bv = ub2[nbase + nt * 16 + lm];
    v8f t = {bv, bv, bv, bv, bv, bv, bv, bv};
    acc2[nt] = t;
  }
  const int ml = mt * 16 + lm;
  {
    ABfrag a[2], bf[2][4];
    load_a_lds(a[0], us, ml, 0, hi);
    load_b4(u2base, 128, bf[0]);
    #pragma unroll
    for (int ks = 0; ks < 4; ++ks) {
      const int cur = ks & 1, nxt = cur ^ 1;
      if (ks + 1 < 4) {
        load_a_lds(a[nxt], us, ml, ks + 1, hi);
        load_b4(u2base + (ks + 1) * 32, 128, bf[nxt]);
      }
      sched_fence();
      acc2[0] = wmma_bf16(a[cur].v, bf[cur][0].v, acc2[0]);
      acc2[1] = wmma_bf16(a[cur].v, bf[cur][1].v, acc2[1]);
      acc2[2] = wmma_bf16(a[cur].v, bf[cur][2].v, acc2[2]);
      acc2[3] = wmma_bf16(a[cur].v, bf[cur][3].v, acc2[3]);
      sched_fence();
    }
  }

  #pragma unroll
  for (int v = 0; v < 8; ++v) {
    long n2 = nb0 + mt * 16 + hi * 8 + v;
    if (n2 < N) {
      #pragma unroll
      for (int nt = 0; nt < 4; ++nt) {
        size_t idx = (size_t)n2 * 128 + nbase + nt * 16 + lm;
        float o = hf[idx] + acc2[nt][v];
        hf[idx] = o;
        hbv[idx] = f32_to_bf16(o);
      }
    }
  }
}

// ---------------------------------------------------------------------------
// Host-side orchestration
// ---------------------------------------------------------------------------
extern "C" void kernel_launch(void* const* d_in, const int* in_sizes, int n_in,
                              void* d_out, int out_size, void* d_ws, size_t ws_size,
                              hipStream_t stream) {
  const int N = 50000, E = 500000;

  const float* h_in  = (const float*)d_in[0];
  const float* x_in  = (const float*)d_in[1];
  const int*   ei    = (const int*)d_in[2];
  const float* eattr = (const float*)d_in[3];
  const int* rows = ei;
  const int* cols = ei + E;

  // params: insertion order — per layer: mW1,mb1,mW2,mb2,aW,ab,uW1,ub1,uW2,ub2
  const float* mW1[2] = {(const float*)d_in[4],  (const float*)d_in[14]};
  const float* mb1[2] = {(const float*)d_in[5],  (const float*)d_in[15]};
  const float* mW2[2] = {(const float*)d_in[6],  (const float*)d_in[16]};
  const float* mb2[2] = {(const float*)d_in[7],  (const float*)d_in[17]};
  const float* aW [2] = {(const float*)d_in[8],  (const float*)d_in[18]};
  const float* ab [2] = {(const float*)d_in[9],  (const float*)d_in[19]};
  const float* uW1[2] = {(const float*)d_in[10], (const float*)d_in[20]};
  const float* ub1[2] = {(const float*)d_in[11], (const float*)d_in[21]};
  const float* uW2[2] = {(const float*)d_in[12], (const float*)d_in[22]};
  const float* ub2[2] = {(const float*)d_in[13], (const float*)d_in[23]};
  const float* cW1 = (const float*)d_in[24];
  const float* cb1 = (const float*)d_in[25];
  const float* cW2 = (const float*)d_in[26];
  const float* cb2 = (const float*)d_in[27];
  const float* cW3 = (const float*)d_in[28];

  // workspace carve-up
  char* ws = (char*)d_ws;
  size_t off = 0;
  auto alloc = [&](size_t bytes) -> void* {
    void* p = ws + off;
    off = (off + bytes + 255) & ~(size_t)255;
    return p;
  };
  float*          hf     = (float*)alloc((size_t)N * 128 * 4);
  unsigned short* hb     = (unsigned short*)alloc((size_t)N * 128 * 2);
  float*          agg    = (float*)alloc((size_t)N * 128 * 4);
  float*          radial = (float*)alloc((size_t)E * 4);
  float*          cdiff  = (float*)alloc((size_t)E * 3 * 4);
  float*          xagg   = (float*)alloc((size_t)N * 3 * 4);
  unsigned short* W1t[2] = {(unsigned short*)alloc(128 * 288 * 2),
                            (unsigned short*)alloc(128 * 288 * 2)};
  unsigned short* W2t[2] = {(unsigned short*)alloc(128 * 128 * 2),
                            (unsigned short*)alloc(128 * 128 * 2)};
  unsigned short* U1t[2] = {(unsigned short*)alloc(128 * 256 * 2),
                            (unsigned short*)alloc(128 * 256 * 2)};
  unsigned short* U2t[2] = {(unsigned short*)alloc(128 * 128 * 2),
                            (unsigned short*)alloc(128 * 128 * 2)};
  unsigned short* C1t = (unsigned short*)alloc(128 * 288 * 2);
  unsigned short* C2t = (unsigned short*)alloc(128 * 128 * 2);

  // prep
  egnn_init_h<<<(N * 128 + 255) / 256, 256, 0, stream>>>(h_in, hf, hb, N * 128);
  egnn_geom<<<(E + 255) / 256, 256, 0, stream>>>(x_in, rows, cols, radial, cdiff, E);
  for (int l = 0; l < 2; ++l) {
    egnn_conv_w<<<(128 * 288 + 255) / 256, 256, 0, stream>>>(mW1[l], W1t[l], 258, 288);
    egnn_conv_w<<<(128 * 128 + 255) / 256, 256, 0, stream>>>(mW2[l], W2t[l], 128, 128);
    egnn_conv_w<<<(128 * 256 + 255) / 256, 256, 0, stream>>>(uW1[l], U1t[l], 256, 256);
    egnn_conv_w<<<(128 * 128 + 255) / 256, 256, 0, stream>>>(uW2[l], U2t[l], 128, 128);
  }
  egnn_conv_w<<<(128 * 288 + 255) / 256, 256, 0, stream>>>(cW1, C1t, 258, 288);
  egnn_conv_w<<<(128 * 128 + 255) / 256, 256, 0, stream>>>(cW2, C2t, 128, 128);

  const int egrid = (E + 63) / 64;
  const int ngrid = (N + 63) / 64;

  for (int l = 0; l < 2; ++l) {
    hipMemsetAsync(agg, 0, (size_t)N * 128 * 4, stream);
    egnn_edge_feat<<<egrid, 256, 0, stream>>>(hb, rows, cols, radial, eattr,
                                              W1t[l], mb1[l], W2t[l], mb2[l],
                                              aW[l], ab[l], agg, E);
    egnn_node<<<ngrid, 256, 0, stream>>>(agg, U1t[l], ub1[l], U2t[l], ub2[l],
                                         hf, hb, N);
  }

  hipMemsetAsync(xagg, 0, (size_t)N * 3 * 4, stream);
  egnn_edge_coord<<<egrid, 256, 0, stream>>>(hb, rows, cols, radial, eattr,
                                             C1t, cb1, C2t, cb2, cW3, cdiff,
                                             xagg, E);

  float* out = (float*)d_out;
  egnn_copy_f32<<<(N * 128 + 255) / 256, 256, 0, stream>>>(hf, out, N * 128);
  egnn_x_final<<<(N * 3 + 255) / 256, 256, 0, stream>>>(x_in, xagg, out + (size_t)N * 128, N * 3);
}
